// NotationGCN_50483045597932
// MI455X (gfx1250) — matrix operand order repaired and checked by reading, not compiled, not run on hardware
//
#include <hip/hip_runtime.h>
#include <hip/hip_bf16.h>
#include <stdint.h>

#define N_NODES 100000
#define N_EDGES 1600000
#define NODE_F  34
#define H       128
#define KP      136   // padded LDS row stride in halfs: 272B => 16B*lane across 16 lanes covers all 64 banks
#define ZPAD    (H * 16)   // offset of the all-zero row in W1eB

typedef __attribute__((ext_vector_type(16))) __bf16       v16bf;
typedef __attribute__((ext_vector_type(8)))  float        v8f;
typedef __attribute__((ext_vector_type(4)))  unsigned int u32x4;
typedef __attribute__((ext_vector_type(4)))  float        f32x4;

union ABreg { v16bf v; u32x4 q[2]; };

__device__ __forceinline__ unsigned short f32_bf16(float f) {
    unsigned int u = __float_as_uint(f);
    u += 0x7FFFu + ((u >> 16) & 1u);          // round-to-nearest-even
    return (unsigned short)(u >> 16);
}

__device__ __forceinline__ v8f splat8(float x) {
    v8f r;
#pragma unroll
    for (int i = 0; i < 8; i++) r[i] = x;
    return r;
}

__device__ __forceinline__ v8f bfwmma(v16bf a, v16bf b, v8f c) {
    return __builtin_amdgcn_wmma_f32_16x16x32_bf16(false, a, false, b, (short)0, c, false, false);
}

// B operand (32x16 KxN, bf16): lane L holds col N=L&15, K-range (L<16 ? 0..15 : 16..31),
// contiguous in K when weights are stored transposed [N][K] in LDS.
__device__ __forceinline__ v16bf ldsB(const unsigned short* p) {
    ABreg r;
    const u32x4* q = (const u32x4*)p;
    r.q[0] = q[0];
    r.q[1] = q[1];
    return r.v;
}

__device__ __forceinline__ float wave_sum(float x) {
#pragma unroll
    for (int o = 16; o > 0; o >>= 1) x += __shfl_xor(x, o, 32);
    return x;
}

// ---------------- node encoder: h = relu(LN(x @ w_enc + b_enc)) ----------------
__global__ void encoder_k(const float* __restrict__ x, const float* __restrict__ w,
                          const float* __restrict__ b, const float* __restrict__ g,
                          const float* __restrict__ be,
                          float* __restrict__ h, unsigned short* __restrict__ hb) {
    int n = blockIdx.x, f = threadIdx.x;
    __shared__ float xr[NODE_F];
    __shared__ float red[8];
    if (f < NODE_F) xr[f] = x[(size_t)n * NODE_F + f];
    __syncthreads();
    float acc = b[f];
#pragma unroll
    for (int k = 0; k < NODE_F; k++) acc += xr[k] * w[k * H + f];
    int wid = threadIdx.x >> 5, lane = threadIdx.x & 31;
    float s = wave_sum(acc);
    if (lane == 0) red[wid] = s;
    __syncthreads();
    float mu = (red[0] + red[1] + red[2] + red[3]) * (1.0f / H);
    float d  = acc - mu;
    float s2 = wave_sum(d * d);
    __syncthreads();
    if (lane == 0) red[wid] = s2;
    __syncthreads();
    float var = (red[0] + red[1] + red[2] + red[3]) * (1.0f / H);
    float y = d * rsqrtf(var + 1e-5f) * g[f] + be[f];
    y = fmaxf(y, 0.0f);
    h[(size_t)n * H + f]  = y;
    hb[(size_t)n * H + f] = f32_bf16(y);
}

// ---------------- degree ----------------
__global__ void zero_k(float* p, int n) {
    int i = blockIdx.x * blockDim.x + threadIdx.x;
    if (i < n) p[i] = 0.0f;
}
__global__ void deg_count_k(const int* __restrict__ tgt, float* deg) {
    int i = blockIdx.x * blockDim.x + threadIdx.x;
    if (i < N_EDGES) atomicAdd(&deg[tgt[i]], 1.0f);
}
__global__ void deg_fin_k(float* dinv, float* deginv) {
    int i = blockIdx.x * blockDim.x + threadIdx.x;
    if (i < N_NODES) {
        float d = dinv[i] + 1.0f;   // deg includes self loop
        dinv[i]   = rsqrtf(d);
        deginv[i] = 1.0f / d;
    }
}

// ---------------- xw = h_bf16 @ W (per-layer GEMM, bf16 WMMA) ----------------
__global__ void gemm_xw_k(const unsigned short* __restrict__ hb, const float* __restrict__ W,
                          float* __restrict__ xw, int n_tiles) {
    __shared__ __align__(16) unsigned short Wt[H * KP];  // transposed [N][K]
    for (int i = threadIdx.x; i < H * H; i += blockDim.x) {
        int k = i >> 7, n = i & 127;
        Wt[n * KP + k] = f32_bf16(W[i]);
    }
    __syncthreads();
    int wid = threadIdx.x >> 5, lane = threadIdx.x & 31;
    int half = lane >> 4, mrow = lane & 15;
    int nwaves = blockDim.x >> 5;
    for (int t = blockIdx.x * nwaves + wid; t < n_tiles; t += gridDim.x * nwaves) {
        // A operand: 16x128 bf16; lane holds row t*16+mrow, K contiguous, 16 per lane half (matches ldsB)
        const u32x4* rp = (const u32x4*)(hb + (size_t)(t * 16 + mrow) * H);
        ABreg a[4];
#pragma unroll
        for (int k = 0; k < 4; k++) {
            a[k].q[0] = rp[4 * k + 2 * half];       // K = k*32 + half*16 .. +7
            a[k].q[1] = rp[4 * k + 2 * half + 1];   // K = k*32 + half*16 + 8 .. +15
        }
        // runtime nt (no unroll) keeps B-operand LDS loads loop-variant -> no LICM hoist/spill
#pragma clang loop unroll(disable)
        for (int nt = 0; nt < 8; nt++) {
            int ncol = nt * 16 + mrow;
            v8f acc = splat8(0.0f);
#pragma unroll
            for (int k = 0; k < 4; k++)
                acc = bfwmma(a[k].v, ldsB(Wt + ncol * KP + k * 32 + half * 16), acc);
            float* op = xw + (size_t)(t * 16 + half * 8) * H + ncol;
#pragma unroll
            for (int v = 0; v < 8; v++) op[(size_t)v * H] = acc[v];   // row = v + half*8
        }
    }
}

// ---------------- agg = deginv*xw + b (self loop + bias init) ----------------
__global__ void self_init_k(const float* __restrict__ xw, const float* __restrict__ deginv,
                            const float* __restrict__ b, float* __restrict__ agg) {
    int i = blockIdx.x * blockDim.x + threadIdx.x;
    if (i >= N_NODES * H) return;
    int n = i >> 7, f = i & 127;
    agg[i] = deginv[n] * xw[i] + b[f];
}

// ---------------- edge scatter: agg[tgt] += dinv[s]*dinv[t]*xw[src] ----------------
__global__ void scatter_k(const int* __restrict__ src, const int* __restrict__ tgt,
                          const float* __restrict__ dinv, const float* __restrict__ xw,
                          float* __restrict__ agg) {
    int e = blockIdx.x * (blockDim.x >> 5) + (threadIdx.x >> 5);
    if (e >= N_EDGES) return;
    int lane = threadIdx.x & 31;
    int s = src[e], t = tgt[e];
    float w = dinv[s] * dinv[t];
    f32x4 x = *((const f32x4*)(xw + (size_t)s * H) + lane);
    float* ap = agg + (size_t)t * H + lane * 4;
    atomicAdd(ap + 0, w * x.x);
    atomicAdd(ap + 1, w * x.y);
    atomicAdd(ap + 2, w * x.z);
    atomicAdd(ap + 3, w * x.w);
}

// ---------------- h = relu(LN(agg)) + h ----------------
__global__ void ln_res_k(const float* __restrict__ agg, const float* __restrict__ g,
                         const float* __restrict__ b, float* __restrict__ h,
                         unsigned short* __restrict__ hb) {
    int n = blockIdx.x, f = threadIdx.x;
    __shared__ float red[8];
    float a = agg[(size_t)n * H + f];
    int wid = threadIdx.x >> 5, lane = threadIdx.x & 31;
    float s = wave_sum(a);
    if (lane == 0) red[wid] = s;
    __syncthreads();
    float mu = (red[0] + red[1] + red[2] + red[3]) * (1.0f / H);
    float d  = a - mu;
    float s2 = wave_sum(d * d);
    __syncthreads();
    if (lane == 0) red[wid] = s2;
    __syncthreads();
    float var = (red[0] + red[1] + red[2] + red[3]) * (1.0f / H);
    float y = d * rsqrtf(var + 1e-5f) * g[f] + b[f];
    y = fmaxf(y, 0.0f) + h[(size_t)n * H + f];
    h[(size_t)n * H + f]  = y;
    hb[(size_t)n * H + f] = f32_bf16(y);
}

// ---------------- fused edge MLP: out = relu(relu([h_s,ea,h_t]@W1+b1)@W2+b2)@w3+b3 ----------------
__global__ void edge_mlp_k(const unsigned short* __restrict__ hb, const int* __restrict__ src,
                           const int* __restrict__ tgt, const float* __restrict__ ea,
                           const float* __restrict__ w1, const float* __restrict__ b1g,
                           const float* __restrict__ w2, const float* __restrict__ b2g,
                           const float* __restrict__ w3g, const float* __restrict__ b3g,
                           float* __restrict__ out, int n_tiles) {
    __shared__ __align__(16) unsigned short W1a[H * KP];       // ec_w1 rows 0..127  (h[src] part), transposed [N][K]
    __shared__ __align__(16) unsigned short W1b[H * KP];       // ec_w1 rows 133..260 (h[tgt] part)
    __shared__ __align__(16) unsigned short W2t[64 * KP];      // ec_w2 transposed [N=64][K=128]
    __shared__ __align__(16) unsigned short W1eB[H * 16 + 16]; // edge_attr part: B layout [N][16], K0..4 valid, + zero row
    __shared__ __align__(16) unsigned short Z1[4][16 * KP];    // per-wave z1 restripe tile
    __shared__ float B1[H], B2[64], W3[64];

    for (int i = threadIdx.x; i < H * H; i += blockDim.x) {
        int k = i >> 7, n = i & 127;
        W1a[n * KP + k] = f32_bf16(w1[k * H + n]);
        W1b[n * KP + k] = f32_bf16(w1[(133 + k) * H + n]);
    }
    for (int i = threadIdx.x; i < H * 64; i += blockDim.x) {
        int k = i >> 6, n = i & 63;
        W2t[n * KP + k] = f32_bf16(w2[i]);
    }
    for (int i = threadIdx.x; i < H * 16 + 16; i += blockDim.x) {
        int n = i >> 4, k = i & 15;
        W1eB[i] = (n < H && k < 5) ? f32_bf16(w1[(H + k) * H + n]) : (unsigned short)0;
    }
    for (int i = threadIdx.x; i < H; i += blockDim.x) B1[i] = b1g[i];
    for (int i = threadIdx.x; i < 64; i += blockDim.x) { B2[i] = b2g[i]; W3[i] = w3g[i]; }
    __syncthreads();
    float b3 = b3g[0];

    int wid = threadIdx.x >> 5, lane = threadIdx.x & 31;
    int half = lane >> 4, mrow = lane & 15;
    int nwaves = blockDim.x >> 5;
    unsigned short* zs = Z1[wid];

    for (int t = blockIdx.x * nwaves + wid; t < n_tiles; t += gridDim.x * nwaves) {
        int e0 = t * 16;
        int sn = src[e0 + mrow], tn = tgt[e0 + mrow];
        const u32x4* ps = (const u32x4*)(hb + (size_t)sn * H);
        const u32x4* pt = (const u32x4*)(hb + (size_t)tn * H);
        ABreg as[4], at4[4];
#pragma unroll
        for (int k = 0; k < 4; k++) {
            as[k].q[0]  = ps[4 * k + 2 * half];
            as[k].q[1]  = ps[4 * k + 2 * half + 1];
            at4[k].q[0] = pt[4 * k + 2 * half];
            at4[k].q[1] = pt[4 * k + 2 * half + 1];
        }
        // edge_attr K=5 padded into a K=32 A operand: lanes 0-15 hold K0..7 (5 valid), upper lanes zero
        ABreg ax;
        ax.q[0][0] = 0u; ax.q[0][1] = 0u; ax.q[0][2] = 0u; ax.q[0][3] = 0u;
        ax.q[1] = ax.q[0];
        if (half == 0) {
            const float* ep = ea + (size_t)(e0 + mrow) * 5;
            ax.q[0][0] = (unsigned)f32_bf16(ep[0]) | ((unsigned)f32_bf16(ep[1]) << 16);
            ax.q[0][1] = (unsigned)f32_bf16(ep[2]) | ((unsigned)f32_bf16(ep[3]) << 16);
            ax.q[0][2] = (unsigned)f32_bf16(ep[4]);
        }
        int eboff = (half == 0) ? 0 : ZPAD;   // upper-half lanes read the zero row
        // ---- layer 1: z1[16][128] ----
#pragma clang loop unroll(disable)
        for (int nt = 0; nt < 8; nt++) {
            int ncol = nt * 16 + mrow;
            v8f acc = splat8(B1[ncol]);
#pragma unroll
            for (int k = 0; k < 4; k++)
                acc = bfwmma(as[k].v, ldsB(W1a + ncol * KP + k * 32 + half * 16), acc);
#pragma unroll
            for (int k = 0; k < 4; k++)
                acc = bfwmma(at4[k].v, ldsB(W1b + ncol * KP + k * 32 + half * 16), acc);
            acc = bfwmma(ax.v, ldsB(W1eB + ((half == 0) ? ncol * 16 : ZPAD)), acc);
#pragma unroll
            for (int v = 0; v < 8; v++) {
                int em = v + half * 8;                       // C layout: row = v + half*8
                zs[em * KP + ncol] = f32_bf16(fmaxf(acc[v], 0.0f));
            }
        }
        (void)eboff;
        asm volatile("s_wait_dscnt 0" ::: "memory");         // per-wave LDS RAW fence
        // ---- layer 2: z2[16][64] + final dot with w3 ----
        const u32x4* pz = (const u32x4*)((const char*)zs + mrow * (KP * 2));
        ABreg az[4];
#pragma unroll
        for (int k = 0; k < 4; k++) {
            az[k].q[0] = pz[4 * k + 2 * half];
            az[k].q[1] = pz[4 * k + 2 * half + 1];
        }
        float rsum[8];
#pragma unroll
        for (int v = 0; v < 8; v++) rsum[v] = 0.0f;
#pragma clang loop unroll(disable)
        for (int nt = 0; nt < 4; nt++) {
            int nc = nt * 16 + mrow;
            v8f acc2 = splat8(B2[nc]);
#pragma unroll
            for (int k = 0; k < 4; k++)
                acc2 = bfwmma(az[k].v, ldsB(W2t + nc * KP + k * 32 + half * 16), acc2);
            float w3v = W3[nc];
#pragma unroll
            for (int v = 0; v < 8; v++) {
                float p = fmaxf(acc2[v], 0.0f) * w3v;
#pragma unroll
                for (int o = 1; o < 16; o <<= 1) p += __shfl_xor(p, o, 32);  // reduce over 16 cols
                rsum[v] += p;
            }
        }
        if (mrow == 0) {
#pragma unroll
            for (int v = 0; v < 8; v++) out[e0 + v + half * 8] = rsum[v] + b3;
        }
    }
}

extern "C" void kernel_launch(void* const* d_in, const int* in_sizes, int n_in,
                              void* d_out, int out_size, void* d_ws, size_t ws_size,
                              hipStream_t stream) {
    (void)in_sizes; (void)n_in; (void)out_size; (void)ws_size;
    const float* x      = (const float*)d_in[0];
    const int*   ei     = (const int*)d_in[1];
    const float* ea     = (const float*)d_in[2];
    const float* w_enc  = (const float*)d_in[3];
    const float* b_enc  = (const float*)d_in[4];
    const float* g_enc  = (const float*)d_in[5];
    const float* be_enc = (const float*)d_in[6];
    const float* gcn_w  = (const float*)d_in[7];
    const float* gcn_b  = (const float*)d_in[8];
    const float* ln_g   = (const float*)d_in[9];
    const float* ln_b   = (const float*)d_in[10];
    const float* ec_w1  = (const float*)d_in[11];
    const float* ec_b1  = (const float*)d_in[12];
    const float* ec_w2  = (const float*)d_in[13];
    const float* ec_b2  = (const float*)d_in[14];
    const float* ec_w3  = (const float*)d_in[15];
    const float* ec_b3  = (const float*)d_in[16];
    const int* src = ei;
    const int* tgt = ei + N_EDGES;

    char* ws = (char*)d_ws;
    float*          h      = (float*)(ws);                       // 51.2 MB
    unsigned short* hbf    = (unsigned short*)(ws + 51200000);   // 25.6 MB
    float*          xw     = (float*)(ws + 76800000);            // 51.2 MB
    float*          agg    = (float*)(ws + 128000000);           // 51.2 MB
    float*          dinv   = (float*)(ws + 179200000);           // 0.4 MB (deg stored here first)
    float*          deginv = (float*)(ws + 179600128);           // 0.4 MB

    encoder_k<<<N_NODES, 128, 0, stream>>>(x, w_enc, b_enc, g_enc, be_enc, h, hbf);

    zero_k<<<(N_NODES + 255) / 256, 256, 0, stream>>>(dinv, N_NODES);
    deg_count_k<<<(N_EDGES + 255) / 256, 256, 0, stream>>>(tgt, dinv);
    deg_fin_k<<<(N_NODES + 255) / 256, 256, 0, stream>>>(dinv, deginv);

    for (int l = 0; l < 3; l++) {
        gemm_xw_k<<<512, 128, 0, stream>>>(hbf, gcn_w + l * H * H, xw, N_NODES / 16);
        self_init_k<<<(N_NODES * H + 255) / 256, 256, 0, stream>>>(xw, deginv, gcn_b + l * H, agg);
        scatter_k<<<N_EDGES / 8, 256, 0, stream>>>(src, tgt, dinv, xw, agg);
        ln_res_k<<<N_NODES, 128, 0, stream>>>(agg, ln_g + l * H, ln_b + l * H, h, hbf);
    }

    edge_mlp_k<<<1024, 128, 0, stream>>>(hbf, src, tgt, ea, ec_w1, ec_b1, ec_w2, ec_b2,
                                         ec_w3, ec_b3, (float*)d_out, N_EDGES / 16);
}
